// CGSL_83674552860819
// MI455X (gfx1250) — matrix operand order repaired and checked by reading, not compiled, not run on hardware
//
#include <hip/hip_runtime.h>

// Problem constants (reference: N=16, L=4096, C=256, NC=512, NR=1, WS=1)
#define NB   16
#define LL   4096
#define CC   256
#define NCC  512

typedef __attribute__((ext_vector_type(16))) __bf16        v16bf;
typedef __attribute__((ext_vector_type(8)))  float         v8f;
typedef __attribute__((ext_vector_type(8)))  unsigned int  v8u;

// ---------- bf16 split helpers (bf16x3 emulated-f32 GEMM) ----------
__device__ __forceinline__ unsigned int f2bf(float f) {
    unsigned int u = __float_as_uint(f);
    u += 0x7FFFu + ((u >> 16) & 1u);      // round-to-nearest-even
    return u >> 16;
}
__device__ __forceinline__ float bf2f(unsigned int h) {
    return __uint_as_float(h << 16);
}
__device__ __forceinline__ void split2(float f0, float f1,
                                       unsigned int& hi, unsigned int& lo) {
    unsigned int h0 = f2bf(f0), h1 = f2bf(f1);
    unsigned int l0 = f2bf(f0 - bf2f(h0));
    unsigned int l1 = f2bf(f1 - bf2f(h1));
    hi = h0 | (h1 << 16);
    lo = l0 | (l1 << 16);
}

// D += Ah*Bh + Ah*Bl + Al*Bh   (16x16x32 bf16 WMMA, wave32)
__device__ __forceinline__ v8f wmma3(v8f c, v8u ah, v8u al, v8u bh, v8u bl) {
    v16bf Ah = __builtin_bit_cast(v16bf, ah);
    v16bf Al = __builtin_bit_cast(v16bf, al);
    v16bf Bh = __builtin_bit_cast(v16bf, bh);
    v16bf Bl = __builtin_bit_cast(v16bf, bl);
    c = __builtin_amdgcn_wmma_f32_16x16x32_bf16(false, Ah, false, Bh, (short)0, c, false, false);
    c = __builtin_amdgcn_wmma_f32_16x16x32_bf16(false, Ah, false, Bl, (short)0, c, false, false);
    c = __builtin_amdgcn_wmma_f32_16x16x32_bf16(false, Al, false, Bh, (short)0, c, false, false);
    return c;
}

// ---------------------------------------------------------------------------
// Kernel 0: pack a row-major f32 matrix (nrows x 256) into two bf16 planes
// (hi, lo), each already in WMMA B-fragment order:
//   plane[((nt*8+ks)*32+lane)*8 + v]
// so a lane's fragment is 8 contiguous u32 (32B) -> direct v8u load.
// ---------------------------------------------------------------------------
__global__ __launch_bounds__(256) void pack_kernel(const float* __restrict__ src,
                                                   unsigned int* __restrict__ dstHi,
                                                   unsigned int* __restrict__ dstLo,
                                                   int nrows)
{
    int p = blockIdx.x * 256 + threadIdx.x;     // global pair id
    if (p >= nrows * 128) return;
    int o     = p >> 7;                         // matrix row (output column)
    int rem   = p & 127;                        // pair within row, k = 2*rem
    int ks    = rem >> 4;
    int r16   = rem & 15;
    int nt    = o >> 4;
    int laneN = o & 15;
    int lane  = ((r16 >> 2) & 1) ? (16 + laneN) : laneN;
    int v     = (r16 & 3) | ((r16 >> 3) << 2);
    unsigned int hi, lo;
    split2(src[o * 256 + 2 * rem], src[o * 256 + 2 * rem + 1], hi, lo);
    int didx = ((nt * 8 + ks) * 32 + lane) * 8 + v;
    dstHi[didx] = hi;
    dstLo[didx] = lo;
}

// A-tile pair -> LDS fragment-order slot.
// Fragment (row, ks, laneHi) holds pairs p = ks*16 + laneHi*4 + {0..3, 8..11}.
__device__ __forceinline__ int apack_idx(int row, int pc) {
    int ks     = pc >> 4;
    int r16    = pc & 15;
    int laneHi = (r16 >> 2) & 1;
    int vv     = (r16 & 3) | ((r16 >> 3) << 2);
    return ((row * 8 + ks) * 2 + laneHi) * 8 + vv;
}

// ---------------------------------------------------------------------------
// Kernel 1: embed = relu(layernorm(inp @ W^T + b) * g + be)
// grid = (2048, 2): 32-row M tile; 8 waves x (2 M-tiles x 2 N-tiles).
// ---------------------------------------------------------------------------
__global__ __launch_bounds__(256) void embed_kernel(
    const float* __restrict__ inp,
    const unsigned int* __restrict__ WxP, const unsigned int* __restrict__ WyP,
    const float* __restrict__ bx, const float* __restrict__ gx, const float* __restrict__ bex,
    const float* __restrict__ by, const float* __restrict__ gy, const float* __restrict__ bey,
    float* __restrict__ xe, float* __restrict__ ye)
{
    const int which = blockIdx.y;
    const unsigned int* WH = which ? WyP : WxP;          // hi plane
    const unsigned int* WL = WH + 256 * 128;             // lo plane
    const float* b  = which ? by  : bx;
    const float* g  = which ? gy  : gx;
    const float* be = which ? bey : bex;
    float* out      = which ? ye  : xe;

    __shared__ unsigned int Ahi[32 * 128];   // fragment-order bf16-hi pairs
    __shared__ unsigned int Alo[32 * 128];
    __shared__ float ldsO[32 * 256];
    __shared__ float sMean[32], sRstd[32];

    const int tid   = threadIdx.x;
    const int mrow0 = blockIdx.x * 32;

    // stage + split the 32x256 activation tile (each element converted once)
    {
        const float2* inp2 = (const float2*)(inp + (size_t)mrow0 * 256);
        #pragma unroll
        for (int i = 0; i < 16; ++i) {
            int p = tid + i * 256;               // pair id 0..4095
            float2 v2 = inp2[p];
            unsigned int hi, lo;
            split2(v2.x, v2.y, hi, lo);
            int d = apack_idx(p >> 7, p & 127);
            Ahi[d] = hi;
            Alo[d] = lo;
        }
    }
    __syncthreads();

    const int wave   = tid >> 5, lane = tid & 31;
    const int laneHi = (lane >> 4) & 1;
    const int row0   = (lane & 15);
    const int row1   = row0 + 16;

    v8f a00 = {}, a01 = {}, a10 = {}, a11 = {};   // [mt][nt]

    for (int ks = 0; ks < 8; ++ks) {
        v8u ah0 = *(const v8u*)&Ahi[((row0 * 8 + ks) * 2 + laneHi) * 8];
        v8u al0 = *(const v8u*)&Alo[((row0 * 8 + ks) * 2 + laneHi) * 8];
        v8u ah1 = *(const v8u*)&Ahi[((row1 * 8 + ks) * 2 + laneHi) * 8];
        v8u al1 = *(const v8u*)&Alo[((row1 * 8 + ks) * 2 + laneHi) * 8];
        #pragma unroll
        for (int t = 0; t < 2; ++t) {
            int nt  = wave * 2 + t;
            size_t off = (size_t)(((nt * 8 + ks) * 32 + lane) * 8);
            v8u bh = *(const v8u*)(WH + off);
            v8u bl = *(const v8u*)(WL + off);
            if (t == 0) { a00 = wmma3(a00, ah0, al0, bh, bl); a10 = wmma3(a10, ah1, al1, bh, bl); }
            else        { a01 = wmma3(a01, ah0, al0, bh, bl); a11 = wmma3(a11, ah1, al1, bh, bl); }
        }
    }

    // accumulators (+bias) -> LDS out tile
    #pragma unroll
    for (int t = 0; t < 2; ++t) {
        int col = (wave * 2 + t) * 16 + (lane & 15);
        float bias = b[col];
        v8f am0 = t ? a01 : a00;
        v8f am1 = t ? a11 : a10;
        #pragma unroll
        for (int j = 0; j < 8; ++j) {
            int r = j + laneHi * 8;
            ldsO[r * 256 + col]        = am0[j] + bias;
            ldsO[(r + 16) * 256 + col] = am1[j] + bias;
        }
    }
    __syncthreads();

    // layernorm stats: 8 consecutive lanes per row
    {
        int r = tid >> 3, cl = tid & 7;
        float s = 0.f, s2 = 0.f;
        #pragma unroll
        for (int j = 0; j < 32; ++j) {
            float x = ldsO[r * 256 + cl + 8 * j];
            s += x; s2 += x * x;
        }
        #pragma unroll
        for (int off = 1; off < 8; off <<= 1) {
            s  += __shfl_xor(s,  off, 32);
            s2 += __shfl_xor(s2, off, 32);
        }
        if (cl == 0) {
            float mean = s * (1.f / 256.f);
            float var  = s2 * (1.f / 256.f) - mean * mean;
            sMean[r] = mean;
            sRstd[r] = rsqrtf(var + 1e-5f);
        }
    }
    __syncthreads();

    #pragma unroll
    for (int i = 0; i < 32; ++i) {
        int idx = tid + i * 256;
        int r = idx >> 8, c = idx & 255;
        float x = ldsO[idx];
        float y = (x - sMean[r]) * sRstd[r] * g[c] + be[c];
        out[(size_t)mrow0 * 256 + idx] = fmaxf(y, 0.f);
    }
}

// ---------------------------------------------------------------------------
// Kernel 2: column sum-of-squares over L.
// ---------------------------------------------------------------------------
__global__ __launch_bounds__(256) void colnorm_kernel(const float* __restrict__ xe,
                                                      float* __restrict__ cn)
{
    const int n = blockIdx.y, chunk = blockIdx.x, c = threadIdx.x;
    const float* base = xe + ((size_t)n * LL + chunk * 128) * CC;
    float s = 0.f;
    for (int l = 0; l < 128; ++l) {
        float v = base[l * CC + c];
        s += v * v;
    }
    atomicAdd(&cn[n * CC + c], s);
}

// ---------------------------------------------------------------------------
// Kernel 3: dists = (x/colnorm) @ means^T, argmax over 512 -> bucket codes.
// grid = 2048 (32-row M tiles); 8 waves x (2 M x 4 N tiles) = 512 columns.
// ---------------------------------------------------------------------------
__global__ __launch_bounds__(256) void dists_kernel(const float* __restrict__ xe,
                                                    const unsigned int* __restrict__ meansP,
                                                    const float* __restrict__ cn,
                                                    int* __restrict__ codes)
{
    __shared__ unsigned int Ahi[32 * 128];
    __shared__ unsigned int Alo[32 * 128];
    __shared__ float ldsD[32 * 512];
    __shared__ float sSc[256];

    const unsigned int* MH = meansP;
    const unsigned int* ML = meansP + 512 * 128;

    const int tid   = threadIdx.x;
    const int mrow0 = blockIdx.x * 32;
    const int n     = mrow0 >> 12;

    {
        float r = sqrtf(cn[n * CC + tid]);
        sSc[tid] = 1.f / fmaxf(r, 1e-12f);
    }
    __syncthreads();

    {
        const float2* xe2 = (const float2*)(xe + (size_t)mrow0 * 256);
        #pragma unroll
        for (int i = 0; i < 16; ++i) {
            int p = tid + i * 256;
            int pc = p & 127;
            float2 v2 = xe2[p];
            unsigned int hi, lo;
            split2(v2.x * sSc[2 * pc], v2.y * sSc[2 * pc + 1], hi, lo);
            int d = apack_idx(p >> 7, pc);
            Ahi[d] = hi;
            Alo[d] = lo;
        }
    }
    __syncthreads();

    const int wave   = tid >> 5, lane = tid & 31;
    const int laneHi = (lane >> 4) & 1;
    const int row0   = (lane & 15);
    const int row1   = row0 + 16;

    v8f acc0[4] = {}, acc1[4] = {};     // [nt] for mt=0 / mt=1

    for (int ks = 0; ks < 8; ++ks) {
        v8u ah0 = *(const v8u*)&Ahi[((row0 * 8 + ks) * 2 + laneHi) * 8];
        v8u al0 = *(const v8u*)&Alo[((row0 * 8 + ks) * 2 + laneHi) * 8];
        v8u ah1 = *(const v8u*)&Ahi[((row1 * 8 + ks) * 2 + laneHi) * 8];
        v8u al1 = *(const v8u*)&Alo[((row1 * 8 + ks) * 2 + laneHi) * 8];
        #pragma unroll
        for (int t = 0; t < 4; ++t) {
            int nt = wave * 4 + t;
            size_t off = (size_t)(((nt * 8 + ks) * 32 + lane) * 8);
            v8u bh = *(const v8u*)(MH + off);
            v8u bl = *(const v8u*)(ML + off);
            acc0[t] = wmma3(acc0[t], ah0, al0, bh, bl);
            acc1[t] = wmma3(acc1[t], ah1, al1, bh, bl);
        }
    }

    #pragma unroll
    for (int t = 0; t < 4; ++t) {
        int col = (wave * 4 + t) * 16 + (lane & 15);
        #pragma unroll
        for (int j = 0; j < 8; ++j) {
            int r = j + laneHi * 8;
            ldsD[r * 512 + col]        = acc0[t][j];
            ldsD[(r + 16) * 512 + col] = acc1[t][j];
        }
    }
    __syncthreads();

    // first-occurrence argmax over 512 per row (8 lanes per row)
    int r = tid >> 3, cl = tid & 7;
    float best = -3.4e38f; int bi = 0;
    for (int j = 0; j < 64; ++j) {
        int c = cl + 8 * j;
        float v = ldsD[r * 512 + c];
        if (v > best) { best = v; bi = c; }
    }
    #pragma unroll
    for (int off = 1; off < 8; off <<= 1) {
        float ov = __shfl_xor(best, off, 32);
        int   oi = __shfl_xor(bi,   off, 32);
        if (ov > best || (ov == best && oi < bi)) { best = ov; bi = oi; }
    }
    if (cl == 0) codes[mrow0 + r] = bi;
}

// ---------------------------------------------------------------------------
// Kernel 4: stable argsort via bitonic sort of (code<<12 | pos) keys in LDS.
// ---------------------------------------------------------------------------
__global__ __launch_bounds__(1024) void sort_kernel(const int* __restrict__ codes,
                                                    int* __restrict__ idxArr)
{
    __shared__ unsigned int keys[LL];
    const int n = blockIdx.x, tid = threadIdx.x;

    for (int i = tid; i < LL; i += 1024)
        keys[i] = (((unsigned)codes[n * LL + i]) << 12) | (unsigned)i;
    __syncthreads();

    for (int k = 2; k <= LL; k <<= 1) {
        for (int j = k >> 1; j > 0; j >>= 1) {
            for (int i = tid; i < LL; i += 1024) {
                int ixj = i ^ j;
                if (ixj > i) {
                    unsigned a = keys[i], c = keys[ixj];
                    bool up = ((i & k) == 0);
                    bool sw = up ? (a > c) : (a < c);
                    if (sw) { keys[i] = c; keys[ixj] = a; }
                }
            }
            __syncthreads();
        }
    }
    for (int i = tid; i < LL; i += 1024)
        idxArr[n * LL + i] = (int)(keys[i] & 4095u);
}

// ---------------------------------------------------------------------------
// Kernel 5: local 3-way attention in sorted order, scattered to d_out.
// ---------------------------------------------------------------------------
__global__ __launch_bounds__(256) void attn_kernel(const float* __restrict__ xe,
                                                   const float* __restrict__ ye,
                                                   const int* __restrict__ idxArr,
                                                   const float* __restrict__ cn,
                                                   float* __restrict__ outF,
                                                   float* __restrict__ bsU)
{
    __shared__ int   sIdx[66];
    __shared__ float sSc[256];

    const int n     = blockIdx.y;
    const int kbase = blockIdx.x * 64;
    const int tid   = threadIdx.x;

    if (tid < 66) {
        int kk = (kbase + tid - 1 + LL) & (LL - 1);
        sIdx[tid] = idxArr[n * LL + kk];
    }
    {
        float r = sqrtf(cn[n * CC + tid]);
        sSc[tid] = 1.f / fmaxf(r, 5e-5f);
    }
    __syncthreads();

    const int wave = tid >> 5, lane = tid & 31;
    const float* xb = xe + (size_t)n * LL * CC;
    const float* yb = ye + (size_t)n * LL * CC;

    for (int q = 0; q < 8; ++q) {
        int koff = wave * 8 + q;
        int i0 = sIdx[koff + 1], im = sIdx[koff], ip = sIdx[koff + 2];
        const float* x0 = xb + i0 * CC;
        const float* xm = xb + im * CC;
        const float* xp = xb + ip * CC;

        float s0 = 0.f, s1 = 0.f, s2 = 0.f;
        #pragma unroll
        for (int j = 0; j < 8; ++j) {
            int c = lane + 32 * j;
            float xv = x0[c], sc = sSc[c];
            s0 += xv * xv * sc;
            s1 += xv * xm[c] * sc;
            s2 += xv * xp[c] * sc;
        }
        #pragma unroll
        for (int off = 1; off < 32; off <<= 1) {
            s0 += __shfl_xor(s0, off, 32);
            s1 += __shfl_xor(s1, off, 32);
            s2 += __shfl_xor(s2, off, 32);
        }
        float m  = fmaxf(s0, fmaxf(s1, s2));
        float e0 = __expf(s0 - m), e1 = __expf(s1 - m), e2 = __expf(s2 - m);
        float Z  = e0 + e1 + e2;
        float inv = 1.f / Z;
        float p0 = e0 * inv, p1 = e1 * inv, p2 = e2 * inv;

        const float* y0 = yb + i0 * CC;
        const float* ym = yb + im * CC;
        const float* yp = yb + ip * CC;
        float* orow = outF + ((size_t)n * LL + i0) * CC;
        #pragma unroll
        for (int j = 0; j < 8; ++j) {
            int c = lane + 32 * j;
            orow[c] = p0 * y0[c] + p1 * ym[c] + p2 * yp[c];
        }
        if (lane == 0) bsU[n * LL + i0] = m + __logf(Z);
    }
}

// ---------------------------------------------------------------------------
// Kernel 6: per-n softmax stats over bucket_score (max, sum of exp)
// ---------------------------------------------------------------------------
__global__ __launch_bounds__(256) void reduce_kernel(const float* __restrict__ bsU,
                                                     float* __restrict__ stats)
{
    __shared__ float red[256];
    const int n = blockIdx.x, tid = threadIdx.x;

    float m = -3.4e38f;
    for (int i = tid; i < LL; i += 256) m = fmaxf(m, bsU[n * LL + i]);
    red[tid] = m; __syncthreads();
    for (int s = 128; s > 0; s >>= 1) {
        if (tid < s) red[tid] = fmaxf(red[tid], red[tid + s]);
        __syncthreads();
    }
    m = red[0]; __syncthreads();

    float z = 0.f;
    for (int i = tid; i < LL; i += 256) z += __expf(bsU[n * LL + i] - m);
    red[tid] = z; __syncthreads();
    for (int s = 128; s > 0; s >>= 1) {
        if (tid < s) red[tid] += red[tid + s];
        __syncthreads();
    }
    if (tid == 0) { stats[2 * n] = m; stats[2 * n + 1] = red[0]; }
}

// ---------------------------------------------------------------------------
// Kernel 7: out = alpha*inp + (1-alpha) * ret * softmax(bs)   (float4)
// ---------------------------------------------------------------------------
__global__ __launch_bounds__(256) void final_kernel(const float* __restrict__ inp,
                                                    const float* __restrict__ bsU,
                                                    const float* __restrict__ stats,
                                                    const float* __restrict__ alphaP,
                                                    float* __restrict__ outF)
{
    int i4   = blockIdx.x * 256 + threadIdx.x;
    int flat = i4 * 4;
    int rowi = flat >> 8;
    int n    = rowi >> 12;

    float alpha = alphaP[0];
    float prob  = __expf(bsU[rowi] - stats[2 * n]) / stats[2 * n + 1];
    float w     = (1.f - alpha) * prob;

    float4 o = ((const float4*)outF)[i4];
    float4 x = ((const float4*)inp)[i4];
    float4 r;
    r.x = alpha * x.x + w * o.x;
    r.y = alpha * x.y + w * o.y;
    r.z = alpha * x.z + w * o.z;
    r.w = alpha * x.w + w * o.w;
    ((float4*)outF)[i4] = r;
}

// ---------------------------------------------------------------------------
extern "C" void kernel_launch(void* const* d_in, const int* in_sizes, int n_in,
                              void* d_out, int out_size, void* d_ws, size_t ws_size,
                              hipStream_t stream)
{
    const float* inp   = (const float*)d_in[0];
    const float* Wx    = (const float*)d_in[1];
    const float* bx    = (const float*)d_in[2];
    const float* gx    = (const float*)d_in[3];
    const float* bex   = (const float*)d_in[4];
    const float* Wy    = (const float*)d_in[5];
    const float* by    = (const float*)d_in[6];
    const float* gy    = (const float*)d_in[7];
    const float* bey   = (const float*)d_in[8];
    const float* means = (const float*)d_in[9];
    const float* alpha = (const float*)d_in[10];
    (void)in_sizes; (void)n_in; (void)out_size; (void)ws_size;

    char* ws = (char*)d_ws;
    const size_t EMB = (size_t)NB * LL * CC * sizeof(float);     // 64 MB
    float*        xe     = (float*)(ws);
    float*        ye     = (float*)(ws + EMB);
    float*        cn     = (float*)(ws + 2 * EMB);
    int*          codes  = (int*)  (ws + 2 * EMB + 65536);
    int*          idxA   = (int*)  (ws + 2 * EMB + 65536 + 262144);
    float*        bsU    = (float*)(ws + 2 * EMB + 65536 + 2 * 262144);
    float*        stats  = (float*)(ws + 2 * EMB + 65536 + 3 * 262144);
    unsigned int* WxP    = (unsigned int*)(ws + 2 * EMB + 65536 + 3 * 262144 + 4096);
    unsigned int* WyP    = WxP + 2 * 256 * 128;                  // hi+lo planes, 256 KB
    unsigned int* meansP = WyP + 2 * 256 * 128;                  // hi+lo planes, 512 KB
    float* outF  = (float*)d_out;

    (void)hipMemsetAsync(cn, 0, NB * CC * sizeof(float), stream);

    pack_kernel   <<<128, 256, 0, stream>>>(Wx, WxP, WxP + 256 * 128, 256);
    pack_kernel   <<<128, 256, 0, stream>>>(Wy, WyP, WyP + 256 * 128, 256);
    pack_kernel   <<<256, 256, 0, stream>>>(means, meansP, meansP + 512 * 128, 512);

    embed_kernel  <<<dim3(2048, 2), 256, 0, stream>>>(inp, WxP, WyP,
                                                      bx, gx, bex, by, gy, bey, xe, ye);
    colnorm_kernel<<<dim3(32, NB), 256, 0, stream>>>(xe, cn);
    dists_kernel  <<<2048, 256, 0, stream>>>(xe, meansP, cn, codes);
    sort_kernel   <<<NB, 1024, 0, stream>>>(codes, idxA);
    attn_kernel   <<<dim3(64, NB), 256, 0, stream>>>(xe, ye, idxA, cn, outF, bsU);
    reduce_kernel <<<NB, 256, 0, stream>>>(bsU, stats);
    final_kernel  <<<16384, 256, 0, stream>>>(inp, bsU, stats, alpha, outF);
}
